// GateLoopBlock_80479097192864
// MI455X (gfx1250) — matrix operand order repaired and checked by reading, not compiled, not run on hardware
//
#include <hip/hip_runtime.h>

typedef __attribute__((ext_vector_type(8)))  __bf16 v8bf;
typedef __attribute__((ext_vector_type(16))) __bf16 v16bf;
typedef __attribute__((ext_vector_type(8)))  float  v8f;
typedef __attribute__((ext_vector_type(4)))  unsigned int v4u;
typedef __attribute__((ext_vector_type(8)))  int v8i;
typedef __attribute__((ext_vector_type(4)))  int v4i;

#if defined(__has_builtin)
#if __has_builtin(__builtin_amdgcn_tensor_load_to_lds) && \
    __has_builtin(__builtin_amdgcn_s_wait_tensorcnt)
#define USE_TDM 1
#endif
#endif
#ifndef USE_TDM
#define USE_TDM 0
#endif

namespace {

constexpr int kS   = 4096;
constexpr int kD   = 1024;
constexpr int kF   = 4096;
constexpr int kTok = 8192;   // B*S
constexpr int kNC  = 32;     // chunks per sequence
constexpr int kLC  = 128;    // chunk length

__device__ __forceinline__ float sigm(float x) { return 1.0f / (1.0f + __expf(-x)); }
__device__ __forceinline__ float gelu_tanh(float x) {
  float x3 = x * x * x;
  return 0.5f * x * (1.0f + tanhf(0.7978845608028654f * (x + 0.044715f * x3)));
}

// ---------------- conversions ----------------
__global__ __launch_bounds__(256) void cvt_f32_bf16_k(const float* __restrict__ in,
                                                      __bf16* __restrict__ out,
                                                      long long n) {
  long long e = (long long)blockIdx.x * 256 + threadIdx.x;
  if (e < n) out[e] = (__bf16)in[e];
}

// W is row-major [K][N] fp32; Wt is [N][K] bf16 (transposed)
__global__ __launch_bounds__(256) void cvt_wt_bf16_k(const float* __restrict__ W,
                                                     __bf16* __restrict__ Wt,
                                                     int K, int N) {
  long long e = (long long)blockIdx.x * 256 + threadIdx.x;
  long long tot = (long long)K * N;
  if (e >= tot) return;
  int n  = (int)(e / K);
  int kk = (int)(e % K);
  Wt[e] = (__bf16)W[(size_t)kk * N + n];
}

#if USE_TDM
// Tensor Data Mover: load a 128-row x 32-elem bf16 tile (rows stride K elems)
// into LDS at lds_addr, with TDM-inserted padding of 4 DWORDs every 16 DWORDs
// => 80B LDS row pitch (matches fragment addressing below).
// D# layout per CDNA5 ISA ch.8 (group0: count/lds/global/type, group1: dims).
__device__ __forceinline__ void tdm_load_tile_128x32(unsigned lds_addr,
                                                     unsigned long long gaddr,
                                                     int K) {
  v4u g0 = {1u,                                   // count=1, user mode
            lds_addr,                             // LDS byte address
            (unsigned)gaddr,                      // global addr [95:64]
            (unsigned)((gaddr >> 32) & 0x01ffffffull) | 0x80000000u};  // addr hi + type=2
  v8i g1 = {(int)((1u << 16)      // data_size = 2 bytes
                  | (1u << 20)    // pad_enable
                  | (3u << 22)    // pad_interval: 16 DWORDs (64B)
                  | (3u << 25)),  // pad_amount: 4 DWORDs (16B)
            (int)(((unsigned)K & 0xffffu) << 16),          // tensor_dim0 lo
            (int)(((unsigned)K >> 16) | (128u << 16)),     // tensor_dim0 hi | tensor_dim1 lo
            (int)(32u << 16),                              // tensor_dim1 hi | tile_dim0=32
            128,                                           // tile_dim1=128, tile_dim2=0
            K,                                             // tensor_dim0_stride lo
            0, 0};                                         // stride hi, dim1_stride
  v4i z4 = {0, 0, 0, 0};
#if __clang_major__ >= 23
  v8i z8 = {0, 0, 0, 0, 0, 0, 0, 0};
  __builtin_amdgcn_tensor_load_to_lds(g0, g1, z4, z4, z8, 0);
#else
  __builtin_amdgcn_tensor_load_to_lds(g0, g1, z4, z4, 0);
#endif
}
#endif

// ---------------- WMMA GEMM: C[M][N] = A[M][K] * Wt[N][K]^T ----------------
// Block: 256 threads = 8 wave32 (2x4 wave grid), 128x128 C tile, K step 32.
// Tile staging via Tensor Data Mover (TDM) when available; LDS rows padded to
// 80B (20-bank stride -> conflict-free ds_load_b128 fragment reads).
__global__ __launch_bounds__(256) void gemm_bf16_wmma(const __bf16* __restrict__ A,
                                                      const __bf16* __restrict__ Wt,
                                                      float* __restrict__ C,
                                                      int M, int N, int K) {
  __shared__ __align__(16) unsigned char smem[4 * 10240];

  const int tid  = threadIdx.x;
  const int wave = tid >> 5;
  const int lane = tid & 31;
  const int wr   = wave >> 2;   // 0..1  (64-row block)
  const int wc   = wave & 3;    // 0..3  (32-col block)
  const int l16  = lane & 15;
  const int hl   = lane >> 4;   // half: 0/1

  const int tile_m0 = blockIdx.y * 128;
  const int tile_n0 = blockIdx.x * 128;
  const int nk = K >> 5;

#if USE_TDM
  // LDS byte address of smem: generic (flat) address low 32 bits are the LDS offset.
  const unsigned lds_base = (unsigned)(unsigned long long)(size_t)(void*)smem;
  const unsigned long long baseA =
      (unsigned long long)(size_t)A + ((size_t)tile_m0 * K) * 2;
  const unsigned long long baseB =
      (unsigned long long)(size_t)Wt + ((size_t)tile_n0 * K) * 2;

  if (tid < 32) {  // wave 0 drives the TDM
    tdm_load_tile_128x32(lds_base, baseA, K);
    tdm_load_tile_128x32(lds_base + 20480, baseB, K);
    __builtin_amdgcn_s_wait_tensorcnt((short)0);
  }
  __syncthreads();
#else
  const int lrow = tid >> 1;
  const int lseg = tid & 1;
  const __bf16* gA = A  + (size_t)(tile_m0 + lrow) * K + lseg * 16;
  const __bf16* gB = Wt + (size_t)(tile_n0 + lrow) * K + lseg * 16;
  const int lds_off = lrow * 80 + lseg * 32;
  {
    uint4 a0 = *(const uint4*)(gA);
    uint4 a1 = *(const uint4*)(gA + 8);
    uint4 b0 = *(const uint4*)(gB);
    uint4 b1 = *(const uint4*)(gB + 8);
    *(uint4*)(smem + lds_off)              = a0;
    *(uint4*)(smem + lds_off + 16)         = a1;
    *(uint4*)(smem + 20480 + lds_off)      = b0;
    *(uint4*)(smem + 20480 + lds_off + 16) = b1;
  }
  __syncthreads();
#endif

  v8f acc[4][2] = {};

  for (int kb = 0; kb < nk; ++kb) {
    const int cur = kb & 1;
    const bool has_next = (kb + 1) < nk;

#if USE_TDM
    if (has_next && tid < 32) {
      const int nxt = cur ^ 1;
      tdm_load_tile_128x32(lds_base + nxt * 10240,
                           baseA + (size_t)(kb + 1) * 64, K);
      tdm_load_tile_128x32(lds_base + 20480 + nxt * 10240,
                           baseB + (size_t)(kb + 1) * 64, K);
      if (kb + 2 < nk) {
        __builtin_prefetch((const void*)(baseA + (size_t)(kb + 2) * 64), 0, 1);
        __builtin_prefetch((const void*)(baseB + (size_t)(kb + 2) * 64), 0, 1);
      }
    }
#else
    uint4 na0, na1, nb0, nb1;
    if (has_next) {
      const __bf16* pa = gA + (size_t)(kb + 1) * 32;
      const __bf16* pb = gB + (size_t)(kb + 1) * 32;
      na0 = *(const uint4*)(pa);
      na1 = *(const uint4*)(pa + 8);
      nb0 = *(const uint4*)(pb);
      nb1 = *(const uint4*)(pb + 8);
      if (kb + 2 < nk) {
        __builtin_prefetch((const void*)(gA + (size_t)(kb + 2) * 32), 0, 1);
        __builtin_prefetch((const void*)(gB + (size_t)(kb + 2) * 32), 0, 1);
      }
    }
#endif

    // A fragments: ISA 16-bit A 16x32 layout: e0..7 -> K=8*hl+e ; e8..15 -> K=16+8*hl+e
    v16bf afrag[4];
#pragma unroll
    for (int mi = 0; mi < 4; ++mi) {
      const unsigned char* p =
          smem + cur * 10240 + (wr * 64 + mi * 16 + l16) * 80 + hl * 16;
      v8bf lo = *(const v8bf*)(p);
      v8bf hi = *(const v8bf*)(p + 32);
      afrag[mi] = __builtin_shufflevector(lo, hi, 0, 1, 2, 3, 4, 5, 6, 7, 8, 9,
                                          10, 11, 12, 13, 14, 15);
    }
    // B fragments: same K order as A: e0..7 -> K=8*hl+e ; e8..15 -> K=16+8*hl+e
    v16bf bfrag[2];
#pragma unroll
    for (int ni = 0; ni < 2; ++ni) {
      const unsigned char* p =
          smem + 20480 + cur * 10240 + (wc * 32 + ni * 16 + l16) * 80 + hl * 16;
      v8bf lo = *(const v8bf*)(p);
      v8bf hi = *(const v8bf*)(p + 32);
      bfrag[ni] = __builtin_shufflevector(lo, hi, 0, 1, 2, 3, 4, 5, 6, 7, 8, 9,
                                          10, 11, 12, 13, 14, 15);
    }

#pragma unroll
    for (int mi = 0; mi < 4; ++mi)
#pragma unroll
      for (int ni = 0; ni < 2; ++ni)
        acc[mi][ni] = __builtin_amdgcn_wmma_f32_16x16x32_bf16(
            false, afrag[mi], false, bfrag[ni], (short)0, acc[mi][ni], false,
            false);

#if USE_TDM
    if (has_next && tid < 32) __builtin_amdgcn_s_wait_tensorcnt((short)0);
#else
    if (has_next) {
      const int nxt = cur ^ 1;
      *(uint4*)(smem + nxt * 10240 + lds_off)              = na0;
      *(uint4*)(smem + nxt * 10240 + lds_off + 16)         = na1;
      *(uint4*)(smem + 20480 + nxt * 10240 + lds_off)      = nb0;
      *(uint4*)(smem + 20480 + nxt * 10240 + lds_off + 16) = nb1;
    }
#endif
    __syncthreads();
  }

  // epilogue: C/D 16x16 f32 layout: VGPR r -> M = r + 8*hl ; N = l16
#pragma unroll
  for (int mi = 0; mi < 4; ++mi) {
#pragma unroll
    for (int ni = 0; ni < 2; ++ni) {
      float* p = C +
                 (size_t)(tile_m0 + wr * 64 + mi * 16 + 8 * hl) * N +
                 (tile_n0 + wc * 32 + ni * 16 + l16);
#pragma unroll
      for (int r = 0; r < 8; ++r) {
        *p = acc[mi][ni][r];
        p += N;
      }
    }
  }
}

// ---------------- gate prep ----------------
// a: [tok][2D] -> complex gate a_c = sigmoid(|a|) * a/|a| (in place, halves)
// k := k*v ; q := q*silu(g)   (all in place)
__global__ __launch_bounds__(256) void gate_prep_k(float* __restrict__ a,
                                                   float* __restrict__ q,
                                                   const float* __restrict__ g,
                                                   float* __restrict__ k,
                                                   const float* __restrict__ v,
                                                   long long n) {
  long long e = (long long)blockIdx.x * 256 + threadIdx.x;
  if (e >= n) return;
  long long tok = e >> 10;
  int d = (int)(e & 1023);
  size_t arow = (size_t)tok * 2048;
  float ar = a[arow + d];
  float ai = a[arow + 1024 + d];
  float mag = sqrtf(ar * ar + ai * ai);
  float s = sigm(mag);
  float arn, ain;
  if (mag > 1e-30f) {
    float sc = s / mag;
    arn = ar * sc;
    ain = ai * sc;
  } else {
    arn = s;  // angle(0)=0 -> e^{i0}=1
    ain = 0.0f;
  }
  a[arow + d] = arn;
  a[arow + 1024 + d] = ain;
  k[e] = k[e] * v[e];
  float gv = g[e];
  q[e] = q[e] * (gv * sigm(gv));
}

// ---------------- chunked complex scan ----------------
__global__ __launch_bounds__(256) void scan_phaseA_k(const float* __restrict__ a,
                                                     const float* __restrict__ kv,
                                                     float* __restrict__ HR,
                                                     float* __restrict__ PR,
                                                     float* __restrict__ PI,
                                                     float4* __restrict__ AGG) {
  int task = blockIdx.x * 256 + threadIdx.x;  // < 2*32*1024 = 65536
  int b = task / (kNC * kD);
  int rem = task % (kNC * kD);
  int c = rem / kD;
  int d = rem % kD;
  float hr = 0.f, hi = 0.f, pr = 1.f, pi = 0.f;
  int t0 = c * kLC;
  for (int j = 0; j < kLC; ++j) {
    int tok = b * kS + t0 + j;
    size_t arow = (size_t)tok * 2048;
    float ar = a[arow + d];
    float ai = a[arow + 1024 + d];
    float kvv = kv[(size_t)tok * kD + d];
    float nhr = ar * hr - ai * hi + kvv;
    float nhi = ai * hr + ar * hi;
    hr = nhr; hi = nhi;
    float npr = ar * pr - ai * pi;
    float npi = ai * pr + ar * pi;
    pr = npr; pi = npi;
    size_t idx = (size_t)tok * kD + d;
    HR[idx] = hr;
    PR[idx] = pr;
    PI[idx] = pi;
  }
  AGG[(size_t)(b * kD + d) * kNC + c] = make_float4(pr, pi, hr, hi);
}

__global__ __launch_bounds__(256) void scan_phaseB_k(const float4* __restrict__ AGG,
                                                     float2* __restrict__ CARRY) {
  int ch = blockIdx.x * 256 + threadIdx.x;  // < 2048
  size_t base = (size_t)ch * kNC;
  float cr = 0.f, ci = 0.f;
  for (int c = 0; c < kNC; ++c) {
    CARRY[base + c] = make_float2(cr, ci);
    float4 g4 = AGG[base + c];
    float ncr = g4.x * cr - g4.y * ci + g4.z;
    float nci = g4.y * cr + g4.x * ci + g4.w;
    cr = ncr; ci = nci;
  }
}

__global__ __launch_bounds__(256) void scan_phaseC_k(const float* __restrict__ QG,
                                                     const float* __restrict__ HR,
                                                     const float* __restrict__ PR,
                                                     const float* __restrict__ PI,
                                                     const float2* __restrict__ CARRY,
                                                     __bf16* __restrict__ yreb,
                                                     long long n) {
  long long e = (long long)blockIdx.x * 256 + threadIdx.x;
  if (e >= n) return;
  long long tok = e >> 10;
  int d = (int)(e & 1023);
  int b = (int)(tok >> 12);
  int t = (int)(tok & 4095);
  int c = t >> 7;
  float2 cy = CARRY[(size_t)(b * kD + d) * kNC + c];
  float hr_tot = PR[e] * cy.x - PI[e] * cy.y + HR[e];
  yreb[e] = (__bf16)(QG[e] * hr_tot);
}

// ---------------- residual + LayerNorm (D=1024, one block per token) ----------------
__global__ __launch_bounds__(256) void ln_res_k(const float* __restrict__ in0,
                                                const float* __restrict__ biasvec,
                                                const float* __restrict__ resid,
                                                const float* __restrict__ sc,
                                                const float* __restrict__ bs,
                                                float* __restrict__ out,
                                                __bf16* __restrict__ outb) {
  const int tok = blockIdx.x;
  const int tid = threadIdx.x;
  const float* pi = in0 + (size_t)tok * kD;
  const float* pr = resid + (size_t)tok * kD;
  float vals[4];
  float s = 0.f, s2 = 0.f;
#pragma unroll
  for (int i = 0; i < 4; ++i) {
    int d = tid + i * 256;
    float v = pi[d] + pr[d];
    if (biasvec) v += biasvec[d];
    vals[i] = v;
    s += v;
    s2 += v * v;
  }
  __shared__ float rs[256], rq[256];
  rs[tid] = s; rq[tid] = s2;
  __syncthreads();
  for (int off = 128; off > 0; off >>= 1) {
    if (tid < off) { rs[tid] += rs[tid + off]; rq[tid] += rq[tid + off]; }
    __syncthreads();
  }
  float mu  = rs[0] * (1.0f / kD);
  float var = rq[0] * (1.0f / kD) - mu * mu;
  float inv = rsqrtf(var + 1e-6f);
#pragma unroll
  for (int i = 0; i < 4; ++i) {
    int d = tid + i * 256;
    float yv = (vals[i] - mu) * inv * sc[d] + bs[d];
    out[(size_t)tok * kD + d] = yv;
    if (outb) outb[(size_t)tok * kD + d] = (__bf16)yv;
  }
}

// ---------------- bias + GELU (tanh) -> bf16 ----------------
__global__ __launch_bounds__(256) void bias_gelu_k(const float* __restrict__ in,
                                                   const float* __restrict__ bias,
                                                   __bf16* __restrict__ out,
                                                   long long n, int ncols) {
  long long e = (long long)blockIdx.x * 256 + threadIdx.x;
  if (e >= n) return;
  int col = (int)(e % ncols);
  out[e] = (__bf16)gelu_tanh(in[e] + bias[col]);
}

}  // namespace

extern "C" void kernel_launch(void* const* d_in, const int* in_sizes, int n_in,
                              void* d_out, int out_size, void* d_ws, size_t ws_size,
                              hipStream_t stream) {
  const float* x    = (const float*)d_in[0];
  const float* Wq   = (const float*)d_in[1];
  const float* Wk   = (const float*)d_in[2];
  const float* Wv   = (const float*)d_in[3];
  const float* Wa   = (const float*)d_in[4];
  const float* Wg   = (const float*)d_in[5];
  const float* Wo   = (const float*)d_in[6];
  const float* ln1s = (const float*)d_in[7];
  const float* ln1b = (const float*)d_in[8];
  const float* W1   = (const float*)d_in[9];
  const float* b1   = (const float*)d_in[10];
  const float* W2   = (const float*)d_in[11];
  const float* b2   = (const float*)d_in[12];
  const float* ln2s = (const float*)d_in[13];
  const float* ln2b = (const float*)d_in[14];

  char* ws = (char*)d_ws;
  constexpr size_t MB = 1ull << 20;
  constexpr size_t EM = 1048576;  // 1M elements

  __bf16* xb   = (__bf16*)(ws + 0);          // 16MB
  __bf16* wb   = (__bf16*)(ws + 16 * MB);    // 30MB bf16 transposed weights
  __bf16* wq_t = wb;
  __bf16* wk_t = wb + 1 * EM;
  __bf16* wv_t = wb + 2 * EM;
  __bf16* wa_t = wb + 3 * EM;  // 2M elems
  __bf16* wg_t = wb + 5 * EM;
  __bf16* wo_t = wb + 6 * EM;
  __bf16* w1_t = wb + 7 * EM;   // 4M elems
  __bf16* w2_t = wb + 11 * EM;  // 4M elems

  float*  qb    = (float*)(ws + 48 * MB);    // 32MB  (becomes q*silu(g))
  float*  kb_   = (float*)(ws + 80 * MB);    // 32MB  (becomes k*v)
  float*  vb    = (float*)(ws + 112 * MB);   // 32MB
  float*  ab    = (float*)(ws + 144 * MB);   // 64MB  [tok][2048], becomes (a.re|a.im)
  float*  gb    = (float*)(ws + 208 * MB);   // 32MB
  float*  HR    = (float*)(ws + 240 * MB);   // 32MB
  float*  PR    = (float*)(ws + 272 * MB);   // 32MB
  float*  PI    = (float*)(ws + 304 * MB);   // 32MB
  float4* AGG   = (float4*)(ws + 336 * MB);  // 1MB
  float2* CARRY = (float2*)(ws + 337 * MB);  // 0.5MB
  __bf16* yreb  = (__bf16*)(ws + 338 * MB);  // 16MB
  float*  attn  = (float*)(ws + 354 * MB);   // 32MB
  float*  y1    = (float*)(ws + 386 * MB);   // 32MB
  __bf16* y1b   = (__bf16*)(ws + 418 * MB);  // 16MB -> 434MB total
  // dead-region reuse:
  float*  mlp1  = (float*)(ws + 48 * MB);    // 128MB over q/k/v/a (dead post-scan)
  __bf16* geluh = (__bf16*)(ws + 208 * MB);  // 64MB over g/HR (dead)
  float*  mlp2  = (float*)(ws + 272 * MB);   // 32MB over PR (dead)

  const int M = kTok;  // 8192
  const long long nMD = (long long)M * kD;   // 8,388,608
  const long long nMF = (long long)M * kF;   // 33,554,432

  // 1. conversions
  cvt_f32_bf16_k<<<dim3((unsigned)((nMD + 255) / 256)), 256, 0, stream>>>(x, xb, nMD);
  cvt_wt_bf16_k<<<dim3((unsigned)((1 * EM + 255) / 256)), 256, 0, stream>>>(Wq, wq_t, kD, kD);
  cvt_wt_bf16_k<<<dim3((unsigned)((1 * EM + 255) / 256)), 256, 0, stream>>>(Wk, wk_t, kD, kD);
  cvt_wt_bf16_k<<<dim3((unsigned)((1 * EM + 255) / 256)), 256, 0, stream>>>(Wv, wv_t, kD, kD);
  cvt_wt_bf16_k<<<dim3((unsigned)((2 * EM + 255) / 256)), 256, 0, stream>>>(Wa, wa_t, kD, 2 * kD);
  cvt_wt_bf16_k<<<dim3((unsigned)((1 * EM + 255) / 256)), 256, 0, stream>>>(Wg, wg_t, kD, kD);
  cvt_wt_bf16_k<<<dim3((unsigned)((1 * EM + 255) / 256)), 256, 0, stream>>>(Wo, wo_t, kD, kD);
  cvt_wt_bf16_k<<<dim3((unsigned)((4 * EM + 255) / 256)), 256, 0, stream>>>(W1, w1_t, kD, kF);
  cvt_wt_bf16_k<<<dim3((unsigned)((4 * EM + 255) / 256)), 256, 0, stream>>>(W2, w2_t, kF, kD);

  // 2. projection GEMMs (WMMA bf16, TDM-staged tiles)
  gemm_bf16_wmma<<<dim3(kD / 128, M / 128), 256, 0, stream>>>(xb, wq_t, qb, M, kD, kD);
  gemm_bf16_wmma<<<dim3(kD / 128, M / 128), 256, 0, stream>>>(xb, wk_t, kb_, M, kD, kD);
  gemm_bf16_wmma<<<dim3(kD / 128, M / 128), 256, 0, stream>>>(xb, wv_t, vb, M, kD, kD);
  gemm_bf16_wmma<<<dim3(2 * kD / 128, M / 128), 256, 0, stream>>>(xb, wa_t, ab, M, 2 * kD, kD);
  gemm_bf16_wmma<<<dim3(kD / 128, M / 128), 256, 0, stream>>>(xb, wg_t, gb, M, kD, kD);

  // 3. gate transform + elementwise products (in place)
  gate_prep_k<<<dim3((unsigned)((nMD + 255) / 256)), 256, 0, stream>>>(ab, qb, gb, kb_, vb, nMD);

  // 4. chunked complex scan (h_t = a_t h_{t-1} + k_t v_t)
  scan_phaseA_k<<<dim3(65536 / 256), 256, 0, stream>>>(ab, kb_, HR, PR, PI, AGG);
  scan_phaseB_k<<<dim3(2048 / 256), 256, 0, stream>>>(AGG, CARRY);
  scan_phaseC_k<<<dim3((unsigned)((nMD + 255) / 256)), 256, 0, stream>>>(qb, HR, PR, PI, CARRY, yreb, nMD);

  // 5. output projection + residual LN1
  gemm_bf16_wmma<<<dim3(kD / 128, M / 128), 256, 0, stream>>>(yreb, wo_t, attn, M, kD, kD);
  ln_res_k<<<dim3(M), 256, 0, stream>>>(attn, nullptr, x, ln1s, ln1b, y1, y1b);

  // 6. MLP
  gemm_bf16_wmma<<<dim3(kF / 128, M / 128), 256, 0, stream>>>(y1b, w1_t, mlp1, M, kF, kD);
  bias_gelu_k<<<dim3((unsigned)((nMF + 255) / 256)), 256, 0, stream>>>(mlp1, b1, geluh, nMF, kF);
  gemm_bf16_wmma<<<dim3(kD / 128, M / 128), 256, 0, stream>>>(geluh, w2_t, mlp2, M, kD, kF);

  // 7. residual LN2 -> output
  ln_res_k<<<dim3(M), 256, 0, stream>>>(mlp2, b2, y1, ln2s, ln2b, (float*)d_out, nullptr);

  (void)in_sizes; (void)n_in; (void)out_size; (void)ws_size;
}
